// SparseConvResBlock_79319456023075
// MI455X (gfx1250) — matrix operand order repaired and checked by reading, not compiled, not run on hardware
//
#include <hip/hip_runtime.h>
#include <hip/hip_bf16.h>

typedef __attribute__((ext_vector_type(16))) __bf16 v16bf;
typedef __attribute__((ext_vector_type(8)))  float  v8f;
typedef unsigned int tdm_v4u __attribute__((ext_vector_type(4)));
typedef int          tdm_v8i __attribute__((ext_vector_type(8)));
typedef int          tdm_v4i __attribute__((ext_vector_type(4)));

#define kB    16
#define kC    256
#define kHW   4096
#define kSKIP 256

__device__ __forceinline__ __bf16 f2bf(float f) {
  union { float f; unsigned u; } v; v.f = f;
  unsigned r = v.u + 0x7FFFu + ((v.u >> 16) & 1u);   // round-to-nearest-even
  unsigned short h = (unsigned short)(r >> 16);
  return __builtin_bit_cast(__bf16, h);
}
__device__ __forceinline__ float bf2f(__bf16 b) {
  unsigned short h = __builtin_bit_cast(unsigned short, b);
  union { unsigned u; float f; } v; v.u = ((unsigned)h) << 16;
  return v.f;
}
__device__ __forceinline__ float gelu_exact(float x) {
  return 0.5f * x * (1.0f + erff(x * 0.70710678118654752f));
}

// ---------------------------------------------------------------------------
// TDM: async 2D tile load, global bf16 row-major [Ntot][Ktot] -> LDS.
// Copies 32 columns starting at gptr for all Ntot rows, padding 1 DWORD every
// 16 DWORDs so the LDS tile lands as [Ntot][34] bf16 (conflict-free reads).
// D# per CDNA5 ISA ch.8 (group0 128b, group1 256b, groups 2/3 zero => 2D).
// ---------------------------------------------------------------------------
__device__ __forceinline__ void tdm_load_tile_b16(
    const __bf16* gptr, unsigned lds_byte_off, unsigned Ktot, unsigned Ntot) {
  unsigned long long ga = (unsigned long long)(size_t)gptr;
  tdm_v4u g0;
  g0[0] = 1u;                                                // count=1
  g0[1] = lds_byte_off;                                      // lds_addr
  g0[2] = (unsigned)(ga & 0xFFFFFFFFu);                      // global_addr lo
  g0[3] = (unsigned)((ga >> 32) & 0x01FFFFFFu) | (2u << 30); // addr hi | type=2
  tdm_v8i g1;
  g1[0] = (int)((1u << 16) | (1u << 20) | (3u << 22));       // 2B, pad 1dw/16dw
  g1[1] = (int)((Ktot & 0xFFFFu) << 16);                     // tensor_dim0 lo16
  g1[2] = (int)(((Ktot >> 16) & 0xFFFFu) | ((Ntot & 0xFFFFu) << 16));
  g1[3] = (int)(((Ntot >> 16) & 0xFFFFu) | (32u << 16));     // tile_dim0=32
  g1[4] = (int)(Ntot & 0xFFFFu);                             // tile_dim1
  g1[5] = (int)Ktot;                                         // dim0_stride lo
  g1[6] = 0;
  g1[7] = 0;
  tdm_v4i gz = {0, 0, 0, 0};
#if defined(__clang_major__) && (__clang_major__ >= 23)
  tdm_v8i gz8 = {0, 0, 0, 0, 0, 0, 0, 0};
  __builtin_amdgcn_tensor_load_to_lds(g0, g1, gz, gz, gz8, 0);
#else
  __builtin_amdgcn_tensor_load_to_lds(g0, g1, gz, gz, 0);
#endif
}

// --- WMMA bf16 fragment packing per CDNA5 ISA 7.12.2 -----------------------
__device__ __forceinline__ v16bf load_fragA(const __bf16* t, int ld, int lane) {
  int m = lane & 15, half = lane >> 4;
  v16bf f;
#pragma unroll
  for (int i = 0; i < 8; ++i) {
    int kb = ((i >> 2) << 4) + (half << 3) + ((i & 3) << 1);
    f[2 * i]     = t[m * ld + kb];
    f[2 * i + 1] = t[m * ld + kb + 1];
  }
  return f;
}
__device__ __forceinline__ v16bf load_fragB(const __bf16* t, int ld, int lane) {
  int n = lane & 15, half = lane >> 4;
  v16bf f;
#pragma unroll
  for (int j = 0; j < 8; ++j) {
    int kb = ((j >> 2) << 4) + (half << 3) + ((j & 3) << 1);
    f[2 * j]     = t[n * ld + kb];
    f[2 * j + 1] = t[n * ld + kb + 1];
  }
  return f;
}

// ===========================================================================
// K0: weight prep: out[N][K] bf16 = transpose(in[K][N] f32)
// ===========================================================================
__global__ void k0_transpose_bf16(const float* __restrict__ in,
                                  __bf16* __restrict__ outp, int Kdim, int Ndim) {
  int idx = blockIdx.x * 256 + threadIdx.x;
  if (idx < Kdim * Ndim) {
    int n = idx / Kdim, k = idx - n * Kdim;
    outp[idx] = f2bf(in[(size_t)k * Ndim + n]);
  }
}

// ===========================================================================
// K1: per-batch modulation vectors
// ===========================================================================
__global__ void k1_modvec(const float* __restrict__ t, const float* __restrict__ z,
    const float* __restrict__ tm1_w, const float* __restrict__ tm1_b,
    const float* __restrict__ tm2_w, const float* __restrict__ tm2_b,
    const float* __restrict__ zm1_w1, const float* __restrict__ zm1_b1,
    const float* __restrict__ zm1_w2, const float* __restrict__ zm1_b2,
    const float* __restrict__ zm2_w1, const float* __restrict__ zm2_b1,
    const float* __restrict__ zm2_w2, const float* __restrict__ zm2_b2,
    float* __restrict__ tt1, float* __restrict__ tt2,
    float* __restrict__ zs1, float* __restrict__ zs2) {
  __shared__ float gt[1024];
  __shared__ float zv[256];
  __shared__ float zh[256];
  const int b = blockIdx.x, tid = threadIdx.x;
  for (int i = tid; i < 1024; i += 256) gt[i] = gelu_exact(t[(size_t)b * 1024 + i]);
  zv[tid] = z[(size_t)b * 256 + tid];
  __syncthreads();
  for (int rep = 0; rep < 2; ++rep) {
    int j = tid + (rep << 8);
    float s1 = tm1_b[j], s2 = tm2_b[j];
    for (int k = 0; k < 1024; ++k) {
      float g = gt[k];
      s1 = fmaf(g, tm1_w[(size_t)k * 512 + j], s1);
      s2 = fmaf(g, tm2_w[(size_t)k * 512 + j], s2);
    }
    tt1[(size_t)b * 512 + j] = s1;
    tt2[(size_t)b * 512 + j] = s2;
  }
  {
    float s = zm1_b1[tid];
    for (int k = 0; k < 256; ++k) s = fmaf(zv[k], zm1_w1[(size_t)k * 256 + tid], s);
    zh[tid] = gelu_exact(s);
  }
  __syncthreads();
  {
    float s = zm1_b2[tid];
    for (int k = 0; k < 256; ++k) s = fmaf(zh[k], zm1_w2[(size_t)k * 256 + tid], s);
    zs1[(size_t)b * 256 + tid] = s;
  }
  __syncthreads();
  {
    float s = zm2_b1[tid];
    for (int k = 0; k < 256; ++k) s = fmaf(zv[k], zm2_w1[(size_t)k * 256 + tid], s);
    zh[tid] = gelu_exact(s);
  }
  __syncthreads();
  {
    float s = zm2_b2[tid];
    for (int k = 0; k < 256; ++k) s = fmaf(zh[k], zm2_w2[(size_t)k * 256 + tid], s);
    zs2[(size_t)b * 256 + tid] = s;
  }
}

// ===========================================================================
// K2: xt = concat(x,skip)^T @ skip_w + b (WMMA; B tiles via TDM)
// ===========================================================================
__global__ void k2_skip_linear_mod1(
    const float* __restrict__ x, const float* __restrict__ skip,
    const __bf16* __restrict__ skip_wT,   // [256 n][512 k] bf16
    const float* __restrict__ skip_b,
    const float* __restrict__ ln_g, const float* __restrict__ ln_b,
    const float* __restrict__ tvec, const float* __restrict__ zvec,
    float* __restrict__ xt, __bf16* __restrict__ hf) {
  extern __shared__ char smem[];
  __bf16* ldsA  = (__bf16*)smem;            // [32][34]
  __bf16* ldsB  = (__bf16*)(smem + 2176);   // [256][34] (TDM destination)
  float*  tileC = (float*)(smem + 19584);   // [32][257]
  float*  stat  = (float*)(smem + 52480);

  const int tid = threadIdx.x, lane = tid & 31, wave = tid >> 5;
  const int bid = blockIdx.x;
  const int b = bid >> 7;
  const int t0 = (bid & 127) << 5;
  const int m_sub = wave & 1;
  const int n_base = (wave >> 1) << 2;

  v8f acc[4];
#pragma unroll
  for (int j = 0; j < 4; ++j)
#pragma unroll
    for (int r = 0; r < 8; ++r) acc[j][r] = 0.0f;

  for (int ks = 0; ks < 512; ks += 32) {
    if (wave == 0)
      tdm_load_tile_b16(skip_wT + ks, 2176u, 512u, 256u);
    const float* src = (ks < 256)
        ? (x    + ((size_t)b * kC    + ks)          * kHW)
        : (skip + ((size_t)b * kSKIP + (ks - 256))  * kHW);
    if (ks + 32 < 512) {
      const float* nsrc = (ks + 32 < 256)
          ? (x    + ((size_t)b * kC    + ks + 32)         * kHW)
          : (skip + ((size_t)b * kSKIP + (ks + 32 - 256)) * kHW);
      __builtin_prefetch(nsrc + (size_t)(tid >> 5) * kHW + t0 + (tid & 31), 0, 1);
    }
    for (int idx = tid; idx < 32 * 32; idx += 256) {
      int ch = idx >> 5, tok = idx & 31;
      ldsA[tok * 34 + ch] = f2bf(src[(size_t)ch * kHW + t0 + tok]);
    }
    if (wave == 0) __builtin_amdgcn_s_wait_tensorcnt(0);
    __syncthreads();
    v16bf a = load_fragA(ldsA + m_sub * 16 * 34, 34, lane);
#pragma unroll
    for (int j = 0; j < 4; ++j) {
      v16bf bfr = load_fragB(ldsB + (n_base + j) * 16 * 34, 34, lane);
      acc[j] = __builtin_amdgcn_wmma_f32_16x16x32_bf16(
          false, a, false, bfr, (short)0, acc[j], false, false);
    }
    __syncthreads();
  }
#pragma unroll
  for (int j = 0; j < 4; ++j) {
    int n0 = (n_base + j) << 4;
#pragma unroll
    for (int r = 0; r < 8; ++r) {
      int m = (m_sub << 4) + r + ((lane >> 4) << 3);
      int n = n0 + (lane & 15);
      tileC[m * 257 + n] = acc[j][r] + skip_b[n];
    }
  }
  __syncthreads();
  if (tid < 32) {
    float s = 0.0f, s2 = 0.0f;
    for (int c = 0; c < 256; ++c) { float v = tileC[tid * 257 + c]; s += v; s2 += v * v; }
    float mean = s * (1.0f / 256.0f);
    float var  = s2 * (1.0f / 256.0f) - mean * mean;
    stat[tid] = mean;
    stat[32 + tid] = rsqrtf(var + 1e-5f);
  }
  __syncthreads();
  const float* ts = tvec + (size_t)b * 512;
  const float* zs = zvec + (size_t)b * 256;
  for (int pass = 0; pass < 32; ++pass) {
    int c = (pass << 3) + (tid >> 5);
    int m = tid & 31;
    float v = tileC[m * 257 + c];
    size_t g = ((size_t)b * kC + c) * kHW + t0 + m;
    xt[g] = v;
    float f = (v - stat[m]) * stat[32 + m] * ln_g[c] + ln_b[c];
    f = f * (1.0f + ts[c]) + ts[256 + c];
    f = f * (1.0f + zs[c]);
    hf[g] = f2bf(f);
  }
}

// ===========================================================================
// K3: depthwise 7x7 SAME conv on hf (bf16 NCHW), /norm, xt += (in place)
// ===========================================================================
__global__ void k3_dwconv_res(
    const __bf16* __restrict__ hf, const float* __restrict__ conv_k,
    const float* __restrict__ norm, float* __restrict__ xt) {
  __shared__ float halo[16 * 22 * 22];
  __shared__ float kwt[16 * 49];
  const int tid = threadIdx.x;
  const int bid = blockIdx.x;
  const int twx = bid & 3, thy = (bid >> 2) & 3, cb = (bid >> 4) & 15, b = bid >> 8;
  const int h0 = thy << 4, w0 = twx << 4, c0 = cb << 4;

  for (int idx = tid; idx < 16 * 22 * 22; idx += 256) {
    int c = idx / 484, rem = idx - c * 484;
    int hy = rem / 22, wx = rem - hy * 22;
    int gh = h0 - 3 + hy, gw = w0 - 3 + wx;
    float v = 0.0f;
    if (gh >= 0 && gh < 64 && gw >= 0 && gw < 64)
      v = bf2f(hf[((size_t)b * kC + c0 + c) * kHW + gh * 64 + gw]);
    halo[idx] = v;
  }
  for (int idx = tid; idx < 16 * 49; idx += 256)
    kwt[idx] = conv_k[(size_t)(c0 + idx / 49) * 49 + idx % 49];
  __syncthreads();
  for (int rep = 0; rep < 16; ++rep) {
    int o = rep * 256 + tid;
    int c = o >> 8, pos = o & 255, oh = pos >> 4, ow = pos & 15;
    float s = 0.0f;
#pragma unroll
    for (int dy = 0; dy < 7; ++dy)
#pragma unroll
      for (int dx = 0; dx < 7; ++dx)
        s = fmaf(halo[(c * 22 + oh + dy) * 22 + ow + dx], kwt[c * 49 + dy * 7 + dx], s);
    int gh = h0 + oh, gw = w0 + ow;
    size_t g = ((size_t)b * kC + c0 + c) * kHW + gh * 64 + gw;
    xt[g] += s / norm[gh * 64 + gw];
  }
}

// ===========================================================================
// K4: LN+FiLM -> MLP (WMMA x2, TDM-fed B tiles) -> residual -> out NCHW
// ===========================================================================
__global__ void k4_mod2_mlp(
    const float* __restrict__ xt,
    const __bf16* __restrict__ w1T,       // [512 n][256 k] bf16
    const float* __restrict__ mb1,
    const __bf16* __restrict__ w2T,       // [256 n][512 k] bf16
    const float* __restrict__ mb2,
    const float* __restrict__ ln_g, const float* __restrict__ ln_b,
    const float* __restrict__ tvec, const float* __restrict__ zvec,
    float* __restrict__ out) {
  extern __shared__ char smem[];
  __bf16* hf2   = (__bf16*)smem;               // [32][258]
  __bf16* G     = (__bf16*)(smem + 16512);     // [32][514]
  __bf16* Bst   = (__bf16*)(smem + 49408);     // [512][34] (TDM destination)
  float*  xtf   = (float*)(smem + 49408);      // phase0 overlay
  float*  outst = (float*)(smem + 16512);      // epilogue overlay
  float*  stat  = (float*)(smem + 84224);

  const int tid = threadIdx.x, lane = tid & 31, wave = tid >> 5;
  const int bid = blockIdx.x;
  const int b = bid >> 7;
  const int t0 = (bid & 127) << 5;
  const int m_sub = wave & 1;

  for (int pass = 0; pass < 32; ++pass) {
    int c = (pass << 3) + (tid >> 5);
    int m = tid & 31;
    xtf[m * 257 + c] = xt[((size_t)b * kC + c) * kHW + t0 + m];
  }
  __syncthreads();
  if (tid < 32) {
    float s = 0.0f, s2 = 0.0f;
    for (int c = 0; c < 256; ++c) { float v = xtf[tid * 257 + c]; s += v; s2 += v * v; }
    float mean = s * (1.0f / 256.0f);
    float var  = s2 * (1.0f / 256.0f) - mean * mean;
    stat[tid] = mean;
    stat[32 + tid] = rsqrtf(var + 1e-5f);
  }
  __syncthreads();
  const float* ts = tvec + (size_t)b * 512;
  const float* zs = zvec + (size_t)b * 256;
  for (int idx = tid; idx < 32 * 256; idx += 256) {
    int m = idx >> 8, c = idx & 255;
    float v = xtf[m * 257 + c];
    float f = (v - stat[m]) * stat[32 + m] * ln_g[c] + ln_b[c];
    f = f * (1.0f + ts[c]) + ts[256 + c];
    f = f * (1.0f + zs[c]);
    hf2[m * 258 + c] = f2bf(f);
  }
  __syncthreads();

  // GEMM1: [32x256] @ [256x512] -> gelu -> G
  {
    const int n_base = (wave >> 1) << 3;
    v8f acc[8];
#pragma unroll
    for (int j = 0; j < 8; ++j)
#pragma unroll
      for (int r = 0; r < 8; ++r) acc[j][r] = 0.0f;
    for (int ks = 0; ks < 256; ks += 32) {
      if (wave == 0) {
        tdm_load_tile_b16(w1T + ks, 49408u, 256u, 512u);
        __builtin_amdgcn_s_wait_tensorcnt(0);
      }
      __syncthreads();
      v16bf a = load_fragA(hf2 + (m_sub << 4) * 258 + ks, 258, lane);
#pragma unroll
      for (int j = 0; j < 8; ++j) {
        v16bf bfr = load_fragB(Bst + (n_base + j) * 16 * 34, 34, lane);
        acc[j] = __builtin_amdgcn_wmma_f32_16x16x32_bf16(
            false, a, false, bfr, (short)0, acc[j], false, false);
      }
      __syncthreads();
    }
#pragma unroll
    for (int j = 0; j < 8; ++j) {
      int n0 = (n_base + j) << 4;
#pragma unroll
      for (int r = 0; r < 8; ++r) {
        int m = (m_sub << 4) + r + ((lane >> 4) << 3);
        int n = n0 + (lane & 15);
        float v = acc[j][r] + mb1[n];
        G[m * 514 + n] = f2bf(gelu_exact(v));
      }
    }
    __syncthreads();
  }

  // GEMM2: [32x512] @ [512x256] -> outst
  {
    const int n_base = (wave >> 1) << 2;
    v8f acc[4];
#pragma unroll
    for (int j = 0; j < 4; ++j)
#pragma unroll
      for (int r = 0; r < 8; ++r) acc[j][r] = 0.0f;
    for (int ks = 0; ks < 512; ks += 32) {
      if (wave == 0) {
        tdm_load_tile_b16(w2T + ks, 49408u, 512u, 256u);
        __builtin_amdgcn_s_wait_tensorcnt(0);
      }
      __syncthreads();
      v16bf a = load_fragA(G + (m_sub << 4) * 514 + ks, 514, lane);
#pragma unroll
      for (int j = 0; j < 4; ++j) {
        v16bf bfr = load_fragB(Bst + (n_base + j) * 16 * 34, 34, lane);
        acc[j] = __builtin_amdgcn_wmma_f32_16x16x32_bf16(
            false, a, false, bfr, (short)0, acc[j], false, false);
      }
      __syncthreads();
    }
#pragma unroll
    for (int j = 0; j < 4; ++j) {
      int n0 = (n_base + j) << 4;
#pragma unroll
      for (int r = 0; r < 8; ++r) {
        int m = (m_sub << 4) + r + ((lane >> 4) << 3);
        int n = n0 + (lane & 15);
        outst[m * 257 + n] = acc[j][r] + mb2[n];
      }
    }
    __syncthreads();
  }

  for (int pass = 0; pass < 32; ++pass) {
    int c = (pass << 3) + (tid >> 5);
    int m = tid & 31;
    size_t g = ((size_t)b * kC + c) * kHW + t0 + m;
    out[g] = outst[m * 257 + c] + xt[g];
  }
}

extern "C" void kernel_launch(void* const* d_in, const int* in_sizes, int n_in,
                              void* d_out, int out_size, void* d_ws, size_t ws_size,
                              hipStream_t stream) {
  (void)in_sizes; (void)n_in; (void)out_size; (void)ws_size;
  const float* x      = (const float*)d_in[0];
  const float* t      = (const float*)d_in[1];
  const float* skip   = (const float*)d_in[2];
  const float* z      = (const float*)d_in[3];
  const float* norm   = (const float*)d_in[4];
  const float* skip_w = (const float*)d_in[5];
  const float* skip_b = (const float*)d_in[6];
  const float* ln1_g  = (const float*)d_in[7];
  const float* ln1_b  = (const float*)d_in[8];
  const float* ln2_g  = (const float*)d_in[9];
  const float* ln2_b  = (const float*)d_in[10];
  const float* tm1_w  = (const float*)d_in[11];
  const float* tm1_b  = (const float*)d_in[12];
  const float* tm2_w  = (const float*)d_in[13];
  const float* tm2_b  = (const float*)d_in[14];
  const float* zm1_w1 = (const float*)d_in[15];
  const float* zm1_b1 = (const float*)d_in[16];
  const float* zm1_w2 = (const float*)d_in[17];
  const float* zm1_b2 = (const float*)d_in[18];
  const float* zm2_w1 = (const float*)d_in[19];
  const float* zm2_b1 = (const float*)d_in[20];
  const float* zm2_w2 = (const float*)d_in[21];
  const float* zm2_b2 = (const float*)d_in[22];
  const float* conv_k = (const float*)d_in[23];
  const float* mlp_w1 = (const float*)d_in[24];
  const float* mlp_b1 = (const float*)d_in[25];
  const float* mlp_w2 = (const float*)d_in[26];
  const float* mlp_b2 = (const float*)d_in[27];
  float* out = (float*)d_out;

  char* w = (char*)d_ws;
  float*  tt1     = (float*)(w);
  float*  tt2     = (float*)(w + (size_t)16 * 512 * 4);
  float*  zs1     = (float*)(w + (size_t)2 * 16 * 512 * 4);
  float*  zs2     = (float*)(w + (size_t)2 * 16 * 512 * 4 + (size_t)16 * 256 * 4);
  __bf16* skip_wT = (__bf16*)(w + 131072);
  __bf16* w1T     = (__bf16*)(w + 131072 + 262144);
  __bf16* w2T     = (__bf16*)(w + 131072 + 524288);
  float*  xtbuf   = (float*)(w + (size_t)(1 << 20));
  __bf16* hf      = (__bf16*)(w + (size_t)(1 << 20) + (size_t)kB * kC * kHW * 4);

  k0_transpose_bf16<<<512, 256, 0, stream>>>(skip_w, skip_wT, 512, 256);
  k0_transpose_bf16<<<512, 256, 0, stream>>>(mlp_w1, w1T, 256, 512);
  k0_transpose_bf16<<<512, 256, 0, stream>>>(mlp_w2, w2T, 512, 256);

  k1_modvec<<<16, 256, 0, stream>>>(t, z, tm1_w, tm1_b, tm2_w, tm2_b,
      zm1_w1, zm1_b1, zm1_w2, zm1_b2, zm2_w1, zm2_b1, zm2_w2, zm2_b2,
      tt1, tt2, zs1, zs2);

  k2_skip_linear_mod1<<<kB * 128, 256, 52736, stream>>>(
      x, skip, skip_wT, skip_b, ln1_g, ln1_b, tt1, zs1, xtbuf, hf);

  k3_dwconv_res<<<4096, 256, 0, stream>>>(hf, conv_k, norm, xtbuf);

  (void)hipFuncSetAttribute((const void*)k4_mod2_mlp,
      hipFuncAttributeMaxDynamicSharedMemorySize, 84480);
  k4_mod2_mlp<<<kB * 128, 256, 84480, stream>>>(
      xtbuf, w1T, mlp_b1, w2T, mlp_b2, ln2_g, ln2_b, tt2, zs2, out);
}
